// GraphCastProcessor_77532749627488
// MI455X (gfx1250) — matrix operand order repaired and checked by reading, not compiled, not run on hardware
//
#include <hip/hip_runtime.h>
#include <hip/hip_bf16.h>

// ---------------- problem constants ----------------
#define NN 40962      // nodes
#define LL 256        // latent
#define EE 4          // edge attr dim
#define DD 163848     // edges
#define NLAYER 6
#define AKE 544       // edge GEMM K (256 + 256 + 4 -> pad to 17*32)
#define AKN 512       // node GEMM1 K (256 + 256)
#define AK2 256       // node GEMM2 K

typedef __bf16  bf16x4  __attribute__((ext_vector_type(4)));
typedef __bf16  bf16x8  __attribute__((ext_vector_type(8)));
typedef __bf16  bf16x16 __attribute__((ext_vector_type(16)));
typedef float   f32x8   __attribute__((ext_vector_type(8)));

__device__ __forceinline__ bf16x16 cat16(bf16x8 a, bf16x8 b) {
    return __builtin_shufflevector(a, b, 0,1,2,3,4,5,6,7,8,9,10,11,12,13,14,15);
}

__device__ __forceinline__ bf16x4 pack4(float4 v) {
    bf16x4 r = { (__bf16)v.x, (__bf16)v.y, (__bf16)v.z, (__bf16)v.w };
    return r;
}

// A fragment: lane (g = lane>>4) holds K = c*32 + g*8 .. +7 and c*32+16+g*8 .. +7 (row M = lane&15)
// B fragment: lane holds K = c*32 + g*8 .. +7 and c*32+16+g*8 .. +7 of column N = lane&15
//   (weights pre-stored column-major so each half is one contiguous 16B run)
template<int STRIDE>
__device__ __forceinline__ void wmma_kchunk(const __bf16* __restrict__ Arow,
                                            const __bf16* __restrict__ Bbase,
                                            int c, int g, f32x8 acc[4]) {
    bf16x16 af = cat16(*(const bf16x8*)(Arow + c*32 + g*8),
                       *(const bf16x8*)(Arow + c*32 + 16 + g*8));
#pragma unroll
    for (int t = 0; t < 4; ++t) {
        const __bf16* Bp = Bbase + (size_t)t*16*STRIDE + c*32 + g*8;
        bf16x16 bf = cat16(*(const bf16x8*)(Bp), *(const bf16x8*)(Bp + 16));
        acc[t] = __builtin_amdgcn_wmma_f32_16x16x32_bf16(
            false, af, false, bf, (short)0, acc[t], false, false);
    }
}

// ---------------- weight prep: fp32 row-major -> bf16 column-major (fragment order) ----------------
__global__ __launch_bounds__(256)
void gc_prep(const float* __restrict__ We_w, const float* __restrict__ Wn1_w,
             const float* __restrict__ Wn2_w,
             __bf16* __restrict__ WeT, __bf16* __restrict__ Wn1T, __bf16* __restrict__ Wn2T) {
    long i = (long)blockIdx.x * 256 + threadIdx.x;
    const long totWe  = (long)NLAYER * 256 * AKE;
    const long totWn1 = (long)NLAYER * 256 * AKN;
    const long totWn2 = (long)NLAYER * 256 * AK2;
    if (i < totWe) {
        long k = i % AKE; long n = (i / AKE) % 256; long l = i / (256L * AKE);
        float v = (k < 2*LL + EE) ? We_w[(l*(2*LL+EE) + k)*LL + n] : 0.0f;
        WeT[i] = (__bf16)v;
        return;
    }
    i -= totWe;
    if (i < totWn1) {
        long k = i % AKN; long n = (i / AKN) % 256; long l = i / (256L * AKN);
        Wn1T[i] = (__bf16)Wn1_w[(l*512 + k)*256 + n];
        return;
    }
    i -= totWn1;
    if (i < totWn2) {
        long k = i % AK2; long n = (i / AK2) % 256; long l = i / (256L * AK2);
        Wn2T[i] = (__bf16)Wn2_w[(l*256 + k)*256 + n];
    }
}

__global__ __launch_bounds__(256)
void gc_zero(float4* __restrict__ p, long n4) {
    long i = (long)blockIdx.x * 256 + threadIdx.x;
    if (i < n4) p[i] = make_float4(0.f, 0.f, 0.f, 0.f);
}

// ---------------- edge kernel: h = [x_i,x_j,ea]@We + b ; silu(LN(h)) ; atomic segment-sum ----------------
__global__ __launch_bounds__(256)
void gc_edge(const float* __restrict__ x, const int* __restrict__ eidx,
             const float* __restrict__ eattr,
             const __bf16* __restrict__ WeT, const float* __restrict__ We_b,
             const float* __restrict__ g1, const float* __restrict__ b1,
             float* __restrict__ agg) {
    // uni: A tile (32 x 544 bf16 = 34816B) aliased later as h tile (32 x 256 f32 = 32768B)
    __shared__ __align__(16) unsigned char uni[32 * AKE * 2];
    __shared__ float redS[256], redQ[256];
    __shared__ float mu[32], rs[32];
    __shared__ int dste[32], srce[32], okv[32];
    __shared__ float web[256], g1l[256], b1l[256];

    const int tid = threadIdx.x;
    const int ebase = blockIdx.x * 32;

    if (tid < 32) {
        int e = ebase + tid;
        int ok = (e < DD) ? 1 : 0;
        int ec = ok ? e : (DD - 1);
        srce[tid] = eidx[ec];        // edge_index[0] = source
        dste[tid] = eidx[DD + ec];   // edge_index[1] = target / aggregation index
        okv[tid] = ok;
    }
    web[tid] = We_b[tid]; g1l[tid] = g1[tid]; b1l[tid] = b1[tid];
    __syncthreads();

    __bf16* Ab = (__bf16*)uni;
    const float4* x4 = (const float4*)x;
    // vectorized gather: 32 rows x 64 float4 chunks each for x_i and x_j
    for (int i = tid; i < 32 * 64; i += 256) {
        int e = i >> 6, n4 = i & 63;
        float4 vi = x4[(size_t)dste[e] * 64 + n4];
        float4 vj = x4[(size_t)srce[e] * 64 + n4];
        *(bf16x4*)(Ab + e * AKE + n4 * 4)       = pack4(vi);  // x_i (target)
        *(bf16x4*)(Ab + e * AKE + 256 + n4 * 4) = pack4(vj);  // x_j (source)
    }
    for (int i = tid; i < 32 * 32; i += 256) {
        int e = i >> 5, k = i & 31;
        int eg = ebase + e; if (eg >= DD) eg = DD - 1;
        Ab[e * AKE + 512 + k] = (k < EE) ? (__bf16)eattr[(size_t)eg * EE + k] : (__bf16)0.0f;
    }
    __syncthreads();

    const int lane = tid & 31, wv = tid >> 5;
    const int mrow = wv >> 2, ncol = wv & 3;
    const int g = lane >> 4, lm = lane & 15;

    f32x8 acc[4] = {};
    const __bf16* Arow  = Ab + (size_t)(mrow * 16 + lm) * AKE;
    const __bf16* Bbase = WeT + (size_t)(ncol * 64 + lm) * AKE;
#pragma unroll 1
    for (int c = 0; c < AKE / 32; ++c)
        wmma_kchunk<AKE>(Arow, Bbase, c, g, acc);

    __syncthreads();                 // everyone done reading A -> reuse as h
    float* hb = (float*)uni;
#pragma unroll
    for (int t = 0; t < 4; ++t)
#pragma unroll
        for (int r = 0; r < 8; ++r) {
            int m = mrow * 16 + g * 8 + r;
            int n = ncol * 64 + t * 16 + lm;
            hb[m * 256 + n] = acc[t][r] + web[n];
        }
    __syncthreads();

    // LayerNorm stats: 8 threads / row, vectorized reads
    {
        int row = tid >> 3, seg = tid & 7;
        float s = 0.f, q = 0.f;
        const float4* hr = (const float4*)(hb + row * 256 + seg * 32);
#pragma unroll
        for (int i = 0; i < 8; ++i) {
            float4 v = hr[i];
            s += v.x + v.y + v.z + v.w;
            q += v.x*v.x + v.y*v.y + v.z*v.z + v.w*v.w;
        }
        redS[tid] = s; redQ[tid] = q;
        __syncthreads();
        if (seg == 0) {
            float S = 0.f, Q = 0.f;
            for (int j = 0; j < 8; ++j) { S += redS[row * 8 + j]; Q += redQ[row * 8 + j]; }
            float m_ = S * (1.f / 256.f);
            float var = Q * (1.f / 256.f) - m_ * m_;
            mu[row] = m_;
            rs[row] = __frsqrt_rn(var + 1e-5f);
        }
        __syncthreads();
    }

    // silu(LN) + atomic segment-sum into agg (4 consecutive cols per thread)
    for (int i = tid * 4; i < 32 * 256; i += 1024) {
        int e = i >> 8, n = i & 255;
        float4 h4 = *(const float4*)(hb + i);
        float m_ = mu[e], r_ = rs[e];
        float v0 = (h4.x - m_) * r_ * g1l[n+0] + b1l[n+0];
        float v1 = (h4.y - m_) * r_ * g1l[n+1] + b1l[n+1];
        float v2 = (h4.z - m_) * r_ * g1l[n+2] + b1l[n+2];
        float v3 = (h4.w - m_) * r_ * g1l[n+3] + b1l[n+3];
        float s0 = v0 * (1.0f / (1.0f + __expf(-v0)));
        float s1 = v1 * (1.0f / (1.0f + __expf(-v1)));
        float s2 = v2 * (1.0f / (1.0f + __expf(-v2)));
        float s3 = v3 * (1.0f / (1.0f + __expf(-v3)));
        if (okv[e]) {
            float* ap = agg + (size_t)dste[e] * 256 + n;
            __hip_atomic_fetch_add(ap + 0, s0, __ATOMIC_RELAXED, __HIP_MEMORY_SCOPE_AGENT);
            __hip_atomic_fetch_add(ap + 1, s1, __ATOMIC_RELAXED, __HIP_MEMORY_SCOPE_AGENT);
            __hip_atomic_fetch_add(ap + 2, s2, __ATOMIC_RELAXED, __HIP_MEMORY_SCOPE_AGENT);
            __hip_atomic_fetch_add(ap + 3, s3, __ATOMIC_RELAXED, __HIP_MEMORY_SCOPE_AGENT);
        }
    }
}

// ---------------- node kernel: u = [x,agg]@Wn1+b ; m = silu(LN(u)) ; out = x + m@Wn2+b ----------------
__global__ __launch_bounds__(256)
void gc_node(const float* __restrict__ xin, const float* __restrict__ agg,
             const __bf16* __restrict__ Wn1T, const float* __restrict__ Wn1_b,
             const float* __restrict__ g2, const float* __restrict__ b2,
             const __bf16* __restrict__ Wn2T, const float* __restrict__ Wn2_b,
             float* __restrict__ xout) {
    __shared__ __align__(16) unsigned char uni[32 * AKN * 2];  // A tile / h tile (both 32KB)
    __shared__ __align__(16) __bf16 Mb[32 * 256];              // silu(LN(u)) as bf16
    __shared__ float redS[256], redQ[256];
    __shared__ float mu[32], rs[32];
    __shared__ int nid[32];
    __shared__ float b1l[256], g2l[256], bt2[256], b2l[256];

    const int tid = threadIdx.x;
    const int nbase = blockIdx.x * 32;

    if (tid < 32) { int nd = nbase + tid; nid[tid] = (nd < NN) ? nd : (NN - 1); }
    b1l[tid] = Wn1_b[tid]; g2l[tid] = g2[tid]; bt2[tid] = b2[tid]; b2l[tid] = Wn2_b[tid];
    __syncthreads();

    __bf16* Ab = (__bf16*)uni;
    const float4* x4 = (const float4*)xin;
    const float4* a4 = (const float4*)agg;
    for (int i = tid; i < 32 * 64; i += 256) {
        int r = i >> 6, n4 = i & 63;
        float4 vx = x4[(size_t)nid[r] * 64 + n4];
        float4 va = a4[(size_t)nid[r] * 64 + n4];
        *(bf16x4*)(Ab + r * AKN + n4 * 4)       = pack4(vx);
        *(bf16x4*)(Ab + r * AKN + 256 + n4 * 4) = pack4(va);
    }
    __syncthreads();

    const int lane = tid & 31, wv = tid >> 5;
    const int mrow = wv >> 2, ncol = wv & 3;
    const int g = lane >> 4, lm = lane & 15;

    // GEMM1: u = [x, agg] @ Wn1
    f32x8 acc[4] = {};
    {
        const __bf16* Arow  = Ab + (size_t)(mrow * 16 + lm) * AKN;
        const __bf16* Bbase = Wn1T + (size_t)(ncol * 64 + lm) * AKN;
#pragma unroll 1
        for (int c = 0; c < AKN / 32; ++c)
            wmma_kchunk<AKN>(Arow, Bbase, c, g, acc);
    }
    __syncthreads();
    float* hb = (float*)uni;
#pragma unroll
    for (int t = 0; t < 4; ++t)
#pragma unroll
        for (int r = 0; r < 8; ++r) {
            int m = mrow * 16 + g * 8 + r;
            int n = ncol * 64 + t * 16 + lm;
            hb[m * 256 + n] = acc[t][r] + b1l[n];
        }
    __syncthreads();

    {
        int row = tid >> 3, seg = tid & 7;
        float s = 0.f, q = 0.f;
        const float4* hr = (const float4*)(hb + row * 256 + seg * 32);
#pragma unroll
        for (int i = 0; i < 8; ++i) {
            float4 v = hr[i];
            s += v.x + v.y + v.z + v.w;
            q += v.x*v.x + v.y*v.y + v.z*v.z + v.w*v.w;
        }
        redS[tid] = s; redQ[tid] = q;
        __syncthreads();
        if (seg == 0) {
            float S = 0.f, Q = 0.f;
            for (int j = 0; j < 8; ++j) { S += redS[row * 8 + j]; Q += redQ[row * 8 + j]; }
            float m_ = S * (1.f / 256.f);
            float var = Q * (1.f / 256.f) - m_ * m_;
            mu[row] = m_;
            rs[row] = __frsqrt_rn(var + 1e-5f);
        }
        __syncthreads();
    }

    // silu(LN(u)) -> bf16 msg tile (4 consecutive cols per thread, packed 8B LDS store)
    for (int i = tid * 4; i < 32 * 256; i += 1024) {
        int r = i >> 8, n = i & 255;
        float4 h4 = *(const float4*)(hb + i);
        float m_ = mu[r], r_ = rs[r];
        float v0 = (h4.x - m_) * r_ * g2l[n+0] + bt2[n+0];
        float v1 = (h4.y - m_) * r_ * g2l[n+1] + bt2[n+1];
        float v2 = (h4.z - m_) * r_ * g2l[n+2] + bt2[n+2];
        float v3 = (h4.w - m_) * r_ * g2l[n+3] + bt2[n+3];
        bf16x4 m4;
        m4[0] = (__bf16)(v0 * (1.0f / (1.0f + __expf(-v0))));
        m4[1] = (__bf16)(v1 * (1.0f / (1.0f + __expf(-v1))));
        m4[2] = (__bf16)(v2 * (1.0f / (1.0f + __expf(-v2))));
        m4[3] = (__bf16)(v3 * (1.0f / (1.0f + __expf(-v3))));
        *(bf16x4*)(Mb + i) = m4;
    }
    __syncthreads();

    // GEMM2: m @ Wn2, residual, write out
    f32x8 acc2[4] = {};
    {
        const __bf16* Arow  = Mb + (size_t)(mrow * 16 + lm) * AK2;
        const __bf16* Bbase = Wn2T + (size_t)(ncol * 64 + lm) * AK2;
#pragma unroll 1
        for (int c = 0; c < AK2 / 32; ++c)
            wmma_kchunk<AK2>(Arow, Bbase, c, g, acc2);
    }
#pragma unroll
    for (int t = 0; t < 4; ++t)
#pragma unroll
        for (int r = 0; r < 8; ++r) {
            int m = mrow * 16 + g * 8 + r;
            int node = nbase + m;
            if (node < NN) {
                int n = ncol * 64 + t * 16 + lm;
                size_t o = (size_t)node * 256 + n;
                xout[o] = xin[o] + acc2[t][r] + b2l[n];
            }
        }
}

// ---------------- host launcher ----------------
extern "C" void kernel_launch(void* const* d_in, const int* in_sizes, int n_in,
                              void* d_out, int out_size, void* d_ws, size_t ws_size,
                              hipStream_t stream) {
    const float* x0    = (const float*)d_in[0];
    const int*   eidx  = (const int*)d_in[1];
    const float* eattr = (const float*)d_in[2];
    const float* We_w  = (const float*)d_in[3];
    const float* We_b  = (const float*)d_in[4];
    const float* g1    = (const float*)d_in[5];
    const float* b1    = (const float*)d_in[6];
    const float* Wn1_w = (const float*)d_in[7];
    const float* Wn1_b = (const float*)d_in[8];
    const float* g2    = (const float*)d_in[9];
    const float* b2    = (const float*)d_in[10];
    const float* Wn2_w = (const float*)d_in[11];
    const float* Wn2_b = (const float*)d_in[12];
    float* out = (float*)d_out;

    char* ws = (char*)d_ws;
    size_t off = 0;
    __bf16* WeT  = (__bf16*)(ws + off); off += (size_t)NLAYER * 256 * AKE * 2;
    __bf16* Wn1T = (__bf16*)(ws + off); off += (size_t)NLAYER * 256 * AKN * 2;
    __bf16* Wn2T = (__bf16*)(ws + off); off += (size_t)NLAYER * 256 * AK2 * 2;
    off = (off + 255) & ~(size_t)255;
    float* xbuf = (float*)(ws + off); off += (size_t)NN * 256 * 4;
    float* agg  = (float*)(ws + off); off += (size_t)NN * 256 * 4;

    {   // weight transpose/convert (once)
        long tot = (long)NLAYER * 256 * (AKE + AKN + AK2);
        int grid = (int)((tot + 255) / 256);
        gc_prep<<<grid, 256, 0, stream>>>(We_w, Wn1_w, Wn2_w, WeT, Wn1T, Wn2T);
    }

    const int EB = (DD + 31) / 32;   // 5121
    const int NB = (NN + 31) / 32;   // 1281
    const long n4 = (long)NN * 256 / 4;
    const int ZG = (int)((n4 + 255) / 256);

    for (int l = 0; l < NLAYER; ++l) {
        const float* xin = (l == 0) ? x0 : xbuf;
        float* xo = (l == NLAYER - 1) ? out : xbuf;   // layers 1..4 update xbuf in place (row-private)
        gc_zero<<<ZG, 256, 0, stream>>>((float4*)agg, n4);
        gc_edge<<<EB, 256, 0, stream>>>(xin, eidx, eattr,
                                        WeT + (size_t)l * 256 * AKE, We_b + l * 256,
                                        g1 + l * 256, b1 + l * 256, agg);
        gc_node<<<NB, 256, 0, stream>>>(xin, agg,
                                        Wn1T + (size_t)l * 256 * AKN, Wn1_b + l * 256,
                                        g2 + l * 256, b2 + l * 256,
                                        Wn2T + (size_t)l * 256 * AK2, Wn2_b + l * 256, xo);
    }
}
